// GRU_37744172597252
// MI455X (gfx1250) — matrix operand order repaired and checked by reading, not compiled, not run on hardware
//
#include <hip/hip_runtime.h>
#include <hip/hip_bf16.h>
#include <cstdint>
#include <cstddef>

// ---------------------------------------------------------------------------
// Problem constants (match reference): T=512, B=64, I=512, H=512
// ---------------------------------------------------------------------------
static constexpr int    kT   = 512;
static constexpr int    kB   = 64;
static constexpr int    kI   = 512;
static constexpr int    kH   = 512;
static constexpr int    kG3H = 3 * kH;                 // 1536
static constexpr size_t kM   = (size_t)kT * kB;        // 32768 rows for gi GEMM

// ---------------------------------------------------------------------------
// WMMA types / helpers (CDNA5 wave32, V_WMMA_F32_16X16X32_BF16)
// ---------------------------------------------------------------------------
typedef __attribute__((ext_vector_type(16))) __bf16       v16bf;
typedef __attribute__((ext_vector_type(8)))  float        v8f;
typedef __attribute__((ext_vector_type(4)))  unsigned int v4u;

union ABf {
    v16bf bf;
    v4u   u4[2];
};

__device__ __forceinline__ unsigned short f2bf(float f) {
    unsigned int u = __float_as_uint(f);
    u += 0x7FFFu + ((u >> 16) & 1u);          // round-to-nearest-even
    return (unsigned short)(u >> 16);
}
__device__ __forceinline__ float bf2f(unsigned short s) {
    return __uint_as_float(((unsigned int)s) << 16);
}
__device__ __forceinline__ float sigmoidf_(float x) {
    return 1.0f / (1.0f + __expf(-x));
}
__device__ __forceinline__ v8f wmma_bf16(v16bf a, v16bf b, v8f c) {
    return __builtin_amdgcn_wmma_f32_16x16x32_bf16(false, a, false, b, (short)0, c,
                                                   false, false);
}

// ---------------------------------------------------------------------------
// Tensor Data Mover support (gfx1250). Guarded so either toolchain compiles.
// ---------------------------------------------------------------------------
#if defined(__has_builtin)
#  if __has_builtin(__builtin_amdgcn_tensor_load_to_lds) && \
      __has_builtin(__builtin_amdgcn_s_wait_tensorcnt)
#    define USE_TDM 1
#  endif
#endif
#ifndef USE_TDM
#  define USE_TDM 0
#endif

#if USE_TDM
typedef __attribute__((ext_vector_type(4))) unsigned int u32x4;
typedef __attribute__((ext_vector_type(8))) int          i32x8;
typedef __attribute__((ext_vector_type(4))) int          i32x4;

// 2D tile DMA: global (row-major, stride_units of 8B) -> packed LDS.
// tile0_units: row length in 8-byte units; tile1_rows: number of rows.
__device__ __forceinline__ void tdm_load_2d(unsigned lds_off, const void* gptr,
                                            unsigned tile0_units,
                                            unsigned tile1_rows,
                                            unsigned long long stride_units) {
    const unsigned long long ga = (unsigned long long)(uintptr_t)gptr;
    u32x4 g0;
    g0[0] = 1u;                                            // count=1, user mode
    g0[1] = lds_off;                                       // lds_addr
    g0[2] = (unsigned)(ga & 0xFFFFFFFFu);                  // global_addr[31:0]
    g0[3] = (unsigned)((ga >> 32) & 0x01FFFFFFu)           // global_addr[56:32]
          | 0x80000000u;                                   // type=2 ("image")

    const unsigned td0 = 0x40000000u, td1 = 0x40000000u;   // huge: no OOB clip
    i32x8 g1;
    g1[0] = (int)(3u << 16);                               // data_size=3 (8B)
    g1[1] = (int)((td0 & 0xFFFFu) << 16);                  // tensor_dim0 lo
    g1[2] = (int)(((td0 >> 16) & 0xFFFFu) | ((td1 & 0xFFFFu) << 16));
    g1[3] = (int)(((td1 >> 16) & 0xFFFFu) | ((tile0_units & 0xFFFFu) << 16));
    g1[4] = (int)(tile1_rows & 0xFFFFu);                   // tile_dim1; tile2=0
    g1[5] = (int)(unsigned)(stride_units & 0xFFFFFFFFull); // dim0_stride lo
    g1[6] = (int)(unsigned)((stride_units >> 32) & 0xFFFFull);
    g1[7] = 0;

    const i32x4 z4 = {0, 0, 0, 0};
#if __clang_major__ >= 23
    const i32x8 z8 = {0, 0, 0, 0, 0, 0, 0, 0};
    __builtin_amdgcn_tensor_load_to_lds(g0, g1, z4, z4, z8, 0);
#else
    __builtin_amdgcn_tensor_load_to_lds(g0, g1, z4, z4, 0);
#endif
}
#endif

// ---------------------------------------------------------------------------
// fp32 -> bf16 conversion
// ---------------------------------------------------------------------------
__global__ void cvt_f32_bf16(const float* __restrict__ in,
                             unsigned short* __restrict__ out, int n) {
    int i = blockIdx.x * blockDim.x + threadIdx.x;
    if (i < n) out[i] = f2bf(in[i]);
}

// ---------------------------------------------------------------------------
// gi(bf16) = A(bf16,[M,K]) * W(bf16,[N,K])^T + b_ih[n] + (n < 2H ? b_hh[n] : 0)
//
// Block: 256 threads = 8 waves, tile 128(M) x 128(N).
// TDM path: double-buffered 128x64 LDS staging of A and W tiles; wave 0
// issues the next chunk's Tensor Data Mover descriptors before computing the
// current chunk, so DMA overlaps the 16 WMMAs per chunk. All 8 B-fragments
// are preloaded before the WMMA burst so dscnt waits stagger.
// ---------------------------------------------------------------------------
__global__ __launch_bounds__(256)
void gemm_gi_bf16(const unsigned short* __restrict__ A,   // [M,K] bf16
                  const unsigned short* __restrict__ W,   // [N,K] bf16
                  const float* __restrict__ bih,          // [N]
                  const float* __restrict__ bhh,          // [N] fold n<2H
                  unsigned short* __restrict__ C,         // [M,N] bf16
                  int K, int N) {
    const int tid  = threadIdx.x;
    const int wave = tid >> 5;
    const int lane = tid & 31;
    const int hi   = lane >> 4;
    const int ln   = lane & 15;

    const int rowBase = blockIdx.x * 128;                 // block M origin
    const int colBase = blockIdx.y * 128;                 // block N origin
    const int myRow   = rowBase + wave * 16;              // wave's 16 M rows

    v8f acc[8];
#pragma unroll
    for (int nt = 0; nt < 8; ++nt) acc[nt] = (v8f){};

#if USE_TDM
    __shared__ unsigned short sA[2][128 * 64];            // 16 KB each
    __shared__ unsigned short sW[2][128 * 64];

    const unsigned long long strideU = (unsigned long long)K / 4;  // 8B units
    auto issue = [&](int buf, int kk) {
        tdm_load_2d((unsigned)(uintptr_t)&sA[buf][0],
                    A + (size_t)rowBase * K + kk, 16u, 128u, strideU);
        tdm_load_2d((unsigned)(uintptr_t)&sW[buf][0],
                    W + (size_t)colBase * K + kk, 16u, 128u, strideU);
    };

    if (tid < 32) issue(0, 0);

    const int KC = K / 64;
    for (int kc = 0; kc < KC; ++kc) {
        const int p = kc & 1;
        if (tid < 32) __builtin_amdgcn_s_wait_tensorcnt((short)0);
        __syncthreads();                 // buf[p] ready; buf[p^1] free
        if (kc + 1 < KC && tid < 32) issue(p ^ 1, (kc + 1) * 64);

#pragma unroll
        for (int ks = 0; ks < 2; ++ks) {
            ABf a;
            const unsigned short* ap =
                &sA[p][(wave * 16 + ln) * 64 + ks * 32 + hi * 16];
            a.u4[0] = *(const v4u*)(ap);
            a.u4[1] = *(const v4u*)(ap + 8);
            ABf b[8];
#pragma unroll
            for (int nt = 0; nt < 8; ++nt) {
                const unsigned short* bp =
                    &sW[p][(nt * 16 + ln) * 64 + ks * 32 + hi * 16];
                b[nt].u4[0] = *(const v4u*)(bp);
                b[nt].u4[1] = *(const v4u*)(bp + 8);
            }
#pragma unroll
            for (int nt = 0; nt < 8; ++nt)
                acc[nt] = wmma_bf16(a.bf, b[nt].bf, acc[nt]);
        }
        __syncthreads();                 // done reading buf[p]
    }
#else
    for (int kk = 0; kk < K; kk += 32) {
        ABf a;
        const unsigned short* ap = A + (size_t)(myRow + ln) * K + kk + hi * 16;
        a.u4[0] = *(const v4u*)(ap);
        a.u4[1] = *(const v4u*)(ap + 8);
        ABf b[8];
#pragma unroll
        for (int nt = 0; nt < 8; ++nt) {
            const unsigned short* bp =
                W + (size_t)(colBase + nt * 16 + ln) * K + kk + hi * 16;
            b[nt].u4[0] = *(const v4u*)(bp);
            b[nt].u4[1] = *(const v4u*)(bp + 8);
        }
#pragma unroll
        for (int nt = 0; nt < 8; ++nt)
            acc[nt] = wmma_bf16(a.bf, b[nt].bf, acc[nt]);
    }
#endif

#pragma unroll
    for (int nt = 0; nt < 8; ++nt) {
        const int   col  = colBase + nt * 16 + ln;
        const float bias = bih[col] + ((col < 2 * kH) ? bhh[col] : 0.0f);
#pragma unroll
        for (int v = 0; v < 8; ++v) {
            const int row = myRow + hi * 8 + v;           // C: M = v + 8*hi
            C[(size_t)row * N + col] = f2bf(acc[nt][v] + bias);
        }
    }
}

// ---------------------------------------------------------------------------
// GRU recurrence, one workgroup per direction (grid.x = dir 0/1).
// 1024 threads = 32 waves; h (bf16, 64 KB) and the current step's gi slice
// (64x1536 bf16, 192 KB) both live in LDS -> 256 KB of the WGP's 320 KB.
// Per step: wave 0 issues ONE contiguous TDM that DMAs gi[t] (192 KB) into
// LDS; the DMA runs concurrently with the whole WMMA phase (gh = h@w_hh^T),
// then s_wait_tensorcnt + the existing barrier publish it for the fused
// gate phase, which now reads gi from LDS instead of 96 global loads/thread.
// ---------------------------------------------------------------------------
__global__ __launch_bounds__(1024)
void gru_scan(const unsigned short* __restrict__ gi_f,
              const unsigned short* __restrict__ gi_b,
              const unsigned short* __restrict__ whh_f,
              const unsigned short* __restrict__ whh_b,
              const float* __restrict__ bhh_f, const float* __restrict__ bhh_b,
              const float* __restrict__ h0,    // [4,B,H] fp32
              int layer,
              unsigned short* __restrict__ y_bf16,  // layer0 output (or null)
              float* __restrict__ y_f32,            // layer1 output (or null)
              float* __restrict__ hn_out) {         // [4,B,H] fp32
    __shared__ unsigned short hbuf[kB * kH];          //  64 KB
#if USE_TDM
    __shared__ unsigned short gibuf[kB * kG3H];       // 192 KB
#endif

    const int dir = blockIdx.x;                       // 0 = fwd, 1 = bwd
    const unsigned short* gi  = dir ? gi_b : gi_f;
    const unsigned short* whh = dir ? whh_b : whh_f;
    const float*          bhhn = (dir ? bhh_b : bhh_f) + 2 * kH;

    const int tid  = threadIdx.x;
    const int wave = tid >> 5;
    const int lane = tid & 31;
    const int hi   = lane >> 4;
    const int ln   = lane & 15;
    const int j0   = wave * 16;                       // this wave's h columns
    const int j    = j0 + ln;

    const size_t h0base = (size_t)(2 * layer + dir) * kB * kH;
    for (int i = tid; i < kB * kH; i += 1024) hbuf[i] = f2bf(h0[h0base + i]);
    __syncthreads();

    const float bnb = bhhn[j];

    for (int s = 0; s < kT; ++s) {
        const int t = dir ? (kT - 1 - s) : s;

#if USE_TDM
        // DMA this step's gi slice (contiguous 192 KB) into LDS; overlaps
        // with the WMMA phase below. Previous step's barrier guarantees the
        // old slice is no longer being read.
        if (tid < 32)
            tdm_load_2d((unsigned)(uintptr_t)&gibuf[0],
                        gi + (size_t)t * kB * kG3H,
                        (unsigned)(kB * kG3H * 2 / 8), 1u,
                        (unsigned long long)(kB * kG3H * 2 / 8));
#endif

        // ---- GEMM phase -------------------------------------------------
        v8f ar[4], az[4], an[4];
#pragma unroll
        for (int m = 0; m < 4; ++m) {
            ar[m] = (v8f){}; az[m] = (v8f){}; an[m] = (v8f){};
        }

#pragma unroll 2
        for (int k = 0; k < kH / 32; ++k) {
            ABf a[4];
#pragma unroll
            for (int m = 0; m < 4; ++m) {
                const unsigned short* hp =
                    &hbuf[(m * 16 + ln) * kH + k * 32 + hi * 16];
                a[m].u4[0] = *(const v4u*)(hp);
                a[m].u4[1] = *(const v4u*)(hp + 8);
            }
            const size_t kofs = (size_t)k * 32 + hi * 16;
            ABf br, bz, bn;
            const unsigned short* wr = whh + (size_t)(j0 + ln) * kH + kofs;
            br.u4[0] = *(const v4u*)(wr);
            br.u4[1] = *(const v4u*)(wr + 8);
            const unsigned short* wz = wr + (size_t)kH * kH;
            bz.u4[0] = *(const v4u*)(wz);
            bz.u4[1] = *(const v4u*)(wz + 8);
            const unsigned short* wn = wz + (size_t)kH * kH;
            bn.u4[0] = *(const v4u*)(wn);
            bn.u4[1] = *(const v4u*)(wn + 8);

#pragma unroll
            for (int m = 0; m < 4; ++m) {
                ar[m] = wmma_bf16(a[m].bf, br.bf, ar[m]);
                az[m] = wmma_bf16(a[m].bf, bz.bf, az[m]);
                an[m] = wmma_bf16(a[m].bf, bn.bf, an[m]);
            }
        }
#if USE_TDM
        if (tid < 32) __builtin_amdgcn_s_wait_tensorcnt((short)0);
#endif
        __syncthreads();   // h reads done; gi slice published

        // ---- fused gate phase: disjoint (b, j) ownership per lane -------
#pragma unroll
        for (int m = 0; m < 4; ++m) {
#pragma unroll
            for (int v = 0; v < 8; ++v) {
                const int brow = m * 16 + hi * 8 + v;
#if USE_TDM
                const float gr = bf2f(gibuf[brow * kG3H + j]);
                const float gz = bf2f(gibuf[brow * kG3H + kH + j]);
                const float gn = bf2f(gibuf[brow * kG3H + 2 * kH + j]);
#else
                const size_t gbase = ((size_t)t * kB + brow) * kG3H + j;
                const float  gr = bf2f(gi[gbase]);
                const float  gz = bf2f(gi[gbase + kH]);
                const float  gn = bf2f(gi[gbase + 2 * kH]);
#endif
                const float r  = sigmoidf_(gr + ar[m][v]);
                const float z  = sigmoidf_(gz + az[m][v]);
                const float n  = tanhf(gn + r * (an[m][v] + bnb));
                const float hp = bf2f(hbuf[brow * kH + j]);
                const float hn = (1.0f - z) * n + z * hp;

                hbuf[brow * kH + j] = f2bf(hn);
                const size_t oidx =
                    ((size_t)t * kB + brow) * (2 * kH) + dir * kH + j;
                if (y_bf16) y_bf16[oidx] = f2bf(hn);
                else        y_f32[oidx]  = hn;
            }
        }
        __syncthreads();   // h fully updated before next step
    }

#pragma unroll
    for (int m = 0; m < 4; ++m)
#pragma unroll
        for (int v = 0; v < 8; ++v) {
            const int brow = m * 16 + hi * 8 + v;
            hn_out[((size_t)(2 * layer + dir) * kB + brow) * kH + j] =
                bf2f(hbuf[brow * kH + j]);
        }
}

// ---------------------------------------------------------------------------
// Host-side orchestration
// ---------------------------------------------------------------------------
extern "C" void kernel_launch(void* const* d_in, const int* in_sizes, int n_in,
                              void* d_out, int out_size, void* d_ws, size_t ws_size,
                              hipStream_t stream) {
    (void)in_sizes; (void)n_in; (void)out_size; (void)ws_size;

    const float* x        = (const float*)d_in[0];
    const float* h0       = (const float*)d_in[1];
    const float* w_ih[4]  = { (const float*)d_in[2],  (const float*)d_in[6],
                              (const float*)d_in[10], (const float*)d_in[14] };
    const float* w_hh[4]  = { (const float*)d_in[3],  (const float*)d_in[7],
                              (const float*)d_in[11], (const float*)d_in[15] };
    const float* b_ih[4]  = { (const float*)d_in[4],  (const float*)d_in[8],
                              (const float*)d_in[12], (const float*)d_in[16] };
    const float* b_hh[4]  = { (const float*)d_in[5],  (const float*)d_in[9],
                              (const float*)d_in[13], (const float*)d_in[17] };

    char* ws = (char*)d_ws;
    constexpr size_t SZ_XBF  = kM * kI * 2;                 //  33.6 MB
    constexpr size_t W_SLOT  = (size_t)kG3H * 1024 * 2;     //   3.1 MB per slot
    constexpr size_t OFF_W   = SZ_XBF;
    constexpr size_t SZ_GI   = kM * (size_t)kG3H * 2;       // 100.7 MB (bf16)
    constexpr size_t OFF_GIF = OFF_W + 8 * W_SLOT;
    constexpr size_t OFF_GIB = OFF_GIF + SZ_GI;
    constexpr size_t OFF_Y0  = OFF_GIB + SZ_GI;             // + 67.1 MB bf16

    unsigned short* xbf  = (unsigned short*)(ws);
    unsigned short* wslt = (unsigned short*)(ws + OFF_W);
    unsigned short* wbf[8];
    for (int i = 0; i < 8; ++i) wbf[i] = wslt + i * (W_SLOT / 2);
    unsigned short* gi_f = (unsigned short*)(ws + OFF_GIF);
    unsigned short* gi_b = (unsigned short*)(ws + OFF_GIB);
    unsigned short* y0bf = (unsigned short*)(ws + OFF_Y0);

    float* out_f32 = (float*)d_out;                          // [T,B,2H]
    float* hn_out  = out_f32 + kM * (size_t)(2 * kH);        // [4,B,H]

    auto cvt = [&](const float* src, unsigned short* dst, int n) {
        cvt_f32_bf16<<<(n + 255) / 256, 256, 0, stream>>>(src, dst, n);
    };
    cvt(x, xbf, (int)(kM * kI));
    const int wih_n[4] = { kG3H * kI, kG3H * kI, kG3H * 2 * kH, kG3H * 2 * kH };
    for (int d = 0; d < 4; ++d) {
        cvt(w_ih[d], wbf[2 * d + 0], wih_n[d]);
        cvt(w_hh[d], wbf[2 * d + 1], kG3H * kH);
    }

    dim3 gGemm((unsigned)(kM / 128), (unsigned)(kG3H / 128));
    gemm_gi_bf16<<<gGemm, 256, 0, stream>>>(xbf, wbf[0], b_ih[0], b_hh[0],
                                            gi_f, kI, kG3H);
    gemm_gi_bf16<<<gGemm, 256, 0, stream>>>(xbf, wbf[2], b_ih[1], b_hh[1],
                                            gi_b, kI, kG3H);
    gru_scan<<<2, 1024, 0, stream>>>(gi_f, gi_b, wbf[1], wbf[3],
                                     b_hh[0], b_hh[1], h0, /*layer=*/0,
                                     y0bf, (float*)nullptr, hn_out);

    gemm_gi_bf16<<<gGemm, 256, 0, stream>>>(y0bf, wbf[4], b_ih[2], b_hh[2],
                                            gi_f, 2 * kH, kG3H);
    gemm_gi_bf16<<<gGemm, 256, 0, stream>>>(y0bf, wbf[6], b_ih[3], b_hh[3],
                                            gi_b, 2 * kH, kG3H);
    gru_scan<<<2, 1024, 0, stream>>>(gi_f, gi_b, wbf[5], wbf[7],
                                     b_hh[2], b_hh[3], h0, /*layer=*/1,
                                     (unsigned short*)nullptr, out_f32, hn_out);
}
